// MultiHeadAttention_34832184770607
// MI455X (gfx1250) — matrix operand order repaired and checked by reading, not compiled, not run on hardware
//
#include <hip/hip_runtime.h>
#include <hip/hip_bf16.h>
#include <math.h>

// Shapes fixed by the reference setup.
#define BATCH 2
#define SEQ   2048
#define DMODEL 2048
#define NHEAD 16
#define DHEAD 128
#define NROWS (BATCH * SEQ)          // 4096
#define NQKV  (3 * DMODEL)           // 6144
#define HEAD_ELEMS ((size_t)BATCH * NHEAD * SEQ * DHEAD)  // 8388608 per buffer

typedef __attribute__((ext_vector_type(16))) __bf16 v16bf;
typedef __attribute__((ext_vector_type(8)))  __bf16 v8bf;
typedef __attribute__((ext_vector_type(8)))  float  v8f;
typedef __attribute__((ext_vector_type(4)))  int    v4i;

// CDNA5 data-mover availability (guarded; safe fallbacks below)
#if defined(__has_builtin)
#if __has_builtin(__builtin_amdgcn_tensor_load_to_lds)
#define HAVE_TDM 1
#endif
#if __has_builtin(__builtin_amdgcn_global_load_async_to_lds_b128)
#define HAVE_ASYNC 1
#endif
#if __has_builtin(__builtin_amdgcn_s_wait_tensorcnt)
#define HAVE_TCNT 1
#endif
#if __has_builtin(__builtin_amdgcn_s_wait_asynccnt)
#define HAVE_ACNT 1
#endif
#endif

typedef __attribute__((address_space(3))) void as3_void;
typedef __attribute__((address_space(1))) v4i  v4i_g;   // global int32x4
typedef __attribute__((address_space(3))) v4i  v4i_l;   // LDS int32x4
typedef __attribute__((ext_vector_type(4))) unsigned tdm_v4u;
typedef __attribute__((ext_vector_type(8))) int      tdm_v8i;
typedef __attribute__((ext_vector_type(4))) int      tdm_v4i;

__device__ __forceinline__ void wait_tensorcnt0() {
#ifdef HAVE_TCNT
  __builtin_amdgcn_s_wait_tensorcnt(0);
#else
  asm volatile("s_wait_tensorcnt 0x0" ::: "memory");
#endif
}

__device__ __forceinline__ void wait_asynccnt0() {
#ifdef HAVE_ACNT
  __builtin_amdgcn_s_wait_asynccnt(0);
#else
  asm volatile("s_wait_asynccnt 0x0" ::: "memory");
#endif
}

#ifdef HAVE_ASYNC
__device__ __forceinline__ void async_copy16(const void* g, void* lds) {
  __builtin_amdgcn_global_load_async_to_lds_b128(
      (v4i_g*)(size_t)g, (v4i_l*)lds, 0, 0);
}
#endif

#ifdef HAVE_TDM
// 2-D TDM load: tile 128(dh) x 32(rows) of bf16, row stride 128 elems,
// LDS padded +8 bf16 per row (pad_interval=5 -> every 64 dwords,
// pad_amount=3 -> 4 dwords). Issue from ONE wave only.
__device__ __forceinline__ void tdm_load_tile_128x32(unsigned long long ga,
                                                     unsigned lds_off) {
  tdm_v4u g0;
  g0[0] = 1u;                                       // count=1 user descriptor
  g0[1] = lds_off;                                  // lds_addr
  g0[2] = (unsigned)ga;                             // global_addr[31:0]
  g0[3] = (unsigned)(ga >> 32) | 0x80000000u;       // addr[56:32] | type=2
  tdm_v8i g1;
  g1[0] = (int)(0x00010000u | (1u << 20) | (5u << 22) | (3u << 25));
  g1[1] = (int)(128u << 16);                        // tensor_dim0 = 128
  g1[2] = (int)(2048u << 16);                       // tensor_dim1 = 2048
  g1[3] = (int)(128u << 16);                        // tile_dim0 = 128
  g1[4] = 32;                                       // tile_dim1 = 32
  g1[5] = 128;                                      // tensor_dim0_stride = 128
  g1[6] = 0;
  g1[7] = 0;
  tdm_v4i gz;
  gz[0] = gz[1] = gz[2] = gz[3] = 0;
#if __clang_major__ >= 23
  tdm_v8i gz8;
#pragma unroll
  for (int i = 0; i < 8; ++i) gz8[i] = 0;
  __builtin_amdgcn_tensor_load_to_lds(g0, g1, gz, gz, gz8, 0);
#else
  __builtin_amdgcn_tensor_load_to_lds(g0, g1, gz, gz, 0);
#endif
}
#endif

__device__ __forceinline__ v8f wmma_bf16(v16bf a, v16bf b, v8f c) {
  return __builtin_amdgcn_wmma_f32_16x16x32_bf16(false, a, false, b, (short)0, c,
                                                 false, false);
}

__device__ __forceinline__ v16bf cat16(v8bf lo, v8bf hi) {
  return __builtin_shufflevector(lo, hi, 0, 1, 2, 3, 4, 5, 6, 7, 8, 9, 10, 11,
                                 12, 13, 14, 15);
}

// Pack 8 f32 -> 8 bf16 (lowers to v_cvt_pk_bf16_f32 x4)
__device__ __forceinline__ v8bf pack8(const float4 f0, const float4 f1) {
  v8bf r;
  r[0] = (__bf16)f0.x; r[1] = (__bf16)f0.y;
  r[2] = (__bf16)f0.z; r[3] = (__bf16)f0.w;
  r[4] = (__bf16)f1.x; r[5] = (__bf16)f1.y;
  r[6] = (__bf16)f1.z; r[7] = (__bf16)f1.w;
  return r;
}

// ---------------------------------------------------------------------------
// Kernel 1: qkv = x @ w_qkv.T  (f32 in, bf16 out scattered to [B,H,T,Dh])
// Block: 256 threads = 8 waves (4x2), 128x128 tile, K-step 32.
// Staging must convert f32->bf16, so it flows through VGPRs (v_cvt_pk).
// ---------------------------------------------------------------------------
__global__ __launch_bounds__(256) void qkv_gemm_kernel(
    const float* __restrict__ x, const float* __restrict__ w_qkv,
    __bf16* __restrict__ qbuf, __bf16* __restrict__ kbuf,
    __bf16* __restrict__ vbuf) {
  __shared__ __bf16 As[128][48];   // [m][k], row stride 96B (16B mult)
  __shared__ __bf16 Bs[128][48];   // [n][k]  (natural w_qkv row layout)

  const int tid  = threadIdx.x;
  const int lane = tid & 31;
  const int wave = tid >> 5;
  const int wm   = wave & 3;       // 4 wave-rows of 32
  const int wn   = wave >> 2;      // 2 wave-cols of 64
  const int row0 = blockIdx.x * 128;
  const int col0 = blockIdx.y * 128;

  const int am     = lane & 15;
  const int koff8  = (lane >> 4) * 8;
  const int koff16 = (lane >> 4) * 16;

  v8f acc[2][4];
#pragma unroll
  for (int tm = 0; tm < 2; ++tm)
#pragma unroll
    for (int tn = 0; tn < 4; ++tn)
#pragma unroll
      for (int j = 0; j < 8; ++j) acc[tm][tn][j] = 0.0f;

  for (int k0 = 0; k0 < DMODEL; k0 += 32) {
    __syncthreads();
#pragma unroll 2
    for (int idx = tid; idx < 512; idx += 256) {
      const int r  = idx >> 2;
      const int c8 = (idx & 3) * 8;
      const float* ap = x + (size_t)(row0 + r) * DMODEL + k0 + c8;
      *(v8bf*)&As[r][c8] =
          pack8(*(const float4*)ap, *(const float4*)(ap + 4));
      const float* bp = w_qkv + (size_t)(col0 + r) * DMODEL + k0 + c8;
      *(v8bf*)&Bs[r][c8] =
          pack8(*(const float4*)bp, *(const float4*)(bp + 4));
    }
    __syncthreads();

    v16bf a[2];
#pragma unroll
    for (int tm = 0; tm < 2; ++tm) {
      const __bf16* ar = &As[wm * 32 + tm * 16 + am][0];
      a[tm] = cat16(*(const v8bf*)(ar + koff16), *(const v8bf*)(ar + koff16 + 8));
    }
#pragma unroll
    for (int tn = 0; tn < 4; ++tn) {
      const __bf16* br = &Bs[wn * 64 + tn * 16 + am][0];
      const v16bf b =
          cat16(*(const v8bf*)(br + koff16), *(const v8bf*)(br + koff16 + 8));
#pragma unroll
      for (int tm = 0; tm < 2; ++tm)
        acc[tm][tn] = wmma_bf16(a[tm], b, acc[tm][tn]);
    }
  }

  // Epilogue: 128-wide column tiles never straddle q/k/v or head boundaries.
  const int which = col0 >> 11;                 // 0=q 1=k 2=v (uniform)
  const int h     = (col0 & 2047) >> 7;         // head (uniform)
  const int bb    = row0 >> 11;                 // batch (uniform)
  __bf16* dst = (which == 0) ? qbuf : (which == 1) ? kbuf : vbuf;
  const size_t head_base = (size_t)(bb * NHEAD + h) * SEQ * DHEAD;

#pragma unroll
  for (int tm = 0; tm < 2; ++tm)
#pragma unroll
    for (int tn = 0; tn < 4; ++tn)
#pragma unroll
      for (int j = 0; j < 8; ++j) {
        const int t  = (row0 + wm * 32 + tm * 16 + j + (lane >> 4) * 8) & 2047;
        const int dh = wn * 64 + tn * 16 + am;
        dst[head_base + ((size_t)t << 7) + dh] = (__bf16)acc[tm][tn][j];
      }
}

// ---------------------------------------------------------------------------
// Kernel 2: in-place RoPE on q,k (bf16). One thread per (b,h,t,j<64) pair.
// ---------------------------------------------------------------------------
__global__ __launch_bounds__(256) void rope_kernel(__bf16* __restrict__ qbuf,
                                                   __bf16* __restrict__ kbuf,
                                                   const int* __restrict__ seq_off) {
  const int gid = blockIdx.x * 256 + threadIdx.x;
  const int j = gid & 63;
  int rest = gid >> 6;
  const int h = rest & 15;
  rest >>= 4;
  const int t = rest & 2047;
  const int b = rest >> 11;
  const size_t base = ((size_t)(b * NHEAD + h) * SEQ + t) << 7;

  const float inv = __powf(10000.0f, -(float)j * (1.0f / 64.0f));
  const float ang = (float)(t + seq_off[0]) * inv;
  const float s = __sinf(ang);
  const float c = __cosf(ang);

  const float q1 = (float)qbuf[base + j];
  const float q2 = (float)qbuf[base + 64 + j];
  qbuf[base + j]      = (__bf16)(q1 * c - q2 * s);
  qbuf[base + 64 + j] = (__bf16)(q2 * c + q1 * s);

  const float k1 = (float)kbuf[base + j];
  const float k2 = (float)kbuf[base + 64 + j];
  kbuf[base + j]      = (__bf16)(k1 * c - k2 * s);
  kbuf[base + 64 + j] = (__bf16)(k2 * c + k1 * s);
}

// ---------------------------------------------------------------------------
// Kernel 3: causal flash attention. Block = (q-tile of 128, b*H+h).
// 8 waves x 16 query rows. K tile DMA'd into LDS by the Tensor Data Mover
// (TENSORcnt); V tile transposed through VGPRs (TDM cannot transpose).
// ---------------------------------------------------------------------------
__global__ __launch_bounds__(256) void attn_kernel(
    const __bf16* __restrict__ qbuf, const __bf16* __restrict__ kbuf,
    const __bf16* __restrict__ vbuf, __bf16* __restrict__ ctx) {
  __shared__ __bf16 Ks[32][136];    // [key][dh], row stride 272B (16B mult)
  __shared__ __bf16 Vt[128][48];    // [dh][key], row stride 96B
  __shared__ __bf16 Ps[8][16][48];  // per-wave P staging, row stride 96B

  const int tid  = threadIdx.x;
  const int lane = tid & 31;
  const int wave = tid >> 5;
  const int bh   = blockIdx.y;
  const int b    = bh >> 4;
  const int h    = bh & 15;
  const int q0   = blockIdx.x * 128;
  const int qrl  = wave * 16;

  const int am     = lane & 15;
  const int koff8  = (lane >> 4) * 8;
  const int koff16 = (lane >> 4) * 16;
  const size_t head_base = (size_t)bh * SEQ * DHEAD;

#ifdef HAVE_TDM
  const unsigned ldsK = (unsigned)(size_t)(as3_void*)&Ks[0][0];
#endif

  // Preload Q fragments (A-layout; two contiguous 16B global loads per chunk)
  v16bf qf[4];
  {
    const __bf16* qrow = qbuf + head_base + (size_t)(q0 + qrl + am) * DHEAD;
#pragma unroll
    for (int kc = 0; kc < 4; ++kc)
      qf[kc] = cat16(*(const v8bf*)(qrow + kc * 32 + koff16),
                     *(const v8bf*)(qrow + kc * 32 + koff16 + 8));
  }

  v8f o[8];
  float mrow[8], lrow[8];
#pragma unroll
  for (int d = 0; d < 8; ++d)
#pragma unroll
    for (int j = 0; j < 8; ++j) o[d][j] = 0.0f;
#pragma unroll
  for (int j = 0; j < 8; ++j) { mrow[j] = -3.0e38f; lrow[j] = 0.0f; }

  const float scale = 0.08838834764831845f;  // 1/sqrt(128)
  const int kv_end = q0 + 128;               // causal bound for this block

  for (int kv0 = 0; kv0 < kv_end; kv0 += 32) {
    __syncthreads();  // previous iteration done reading LDS

    // ---- K tile: TDM if available, else async copies, else sync copies ----
#ifdef HAVE_TDM
    if (tid < 32)  // one wave issues the DMA (EXEC-independent, wave-level op)
      tdm_load_tile_128x32(
          (unsigned long long)(size_t)(kbuf + head_base +
                                       (size_t)kv0 * DHEAD),
          ldsK);
#else
#pragma unroll 2
    for (int idx = tid; idx < 512; idx += 256) {
      const int r = idx >> 4;
      const int c = (idx & 15) * 8;
      const __bf16* src = kbuf + head_base + (size_t)(kv0 + r) * DHEAD + c;
#ifdef HAVE_ASYNC
      async_copy16(src, &Ks[r][c]);
#else
      *(uint4*)&Ks[r][c] = *(const uint4*)src;
#endif
    }
#endif

    // ---- V tile: register transpose into Vt[dh][key] ----
#pragma unroll 2
    for (int idx = tid; idx < 512; idx += 256) {
      const int r = idx >> 4;        // key row
      const int c = (idx & 15) * 8;  // dh base
      const v8bf vv =
          *(const v8bf*)(vbuf + head_base + (size_t)(kv0 + r) * DHEAD + c);
#pragma unroll
      for (int jj = 0; jj < 8; ++jj) Vt[c + jj][r] = vv[jj];
    }

#ifdef HAVE_TDM
    wait_tensorcnt0();
#elif defined(HAVE_ASYNC)
    wait_asynccnt0();
#endif
    __syncthreads();

    // S = Q K^T for two 16-key subtiles
    v8f s[2];
#pragma unroll
    for (int sub = 0; sub < 2; ++sub) {
#pragma unroll
      for (int j = 0; j < 8; ++j) s[sub][j] = 0.0f;
      const __bf16* kr = &Ks[sub * 16 + am][0];
#pragma unroll
      for (int kc = 0; kc < 4; ++kc) {
        const v16bf kb = cat16(*(const v8bf*)(kr + kc * 32 + koff16),
                               *(const v8bf*)(kr + kc * 32 + koff16 + 8));
        s[sub] = wmma_bf16(qf[kc], kb, s[sub]);
      }
    }

    // Online softmax (row values live in one 16-lane half; reduce via xor)
#pragma unroll
    for (int j = 0; j < 8; ++j) {
      const int qrow = q0 + qrl + j + (lane >> 4) * 8;
      float s0 = s[0][j] * scale;
      float s1 = s[1][j] * scale;
      if (kv0 + am > qrow)      s0 = -3.0e38f;
      if (kv0 + 16 + am > qrow) s1 = -3.0e38f;
      float v = fmaxf(s0, s1);
      v = fmaxf(v, __shfl_xor(v, 1, 32));
      v = fmaxf(v, __shfl_xor(v, 2, 32));
      v = fmaxf(v, __shfl_xor(v, 4, 32));
      v = fmaxf(v, __shfl_xor(v, 8, 32));
      const float mnew = fmaxf(mrow[j], v);
      const float corr = __expf(mrow[j] - mnew);
      const float p0 = __expf(s0 - mnew);
      const float p1 = __expf(s1 - mnew);
      float rs = p0 + p1;
      rs += __shfl_xor(rs, 1, 32);
      rs += __shfl_xor(rs, 2, 32);
      rs += __shfl_xor(rs, 4, 32);
      rs += __shfl_xor(rs, 8, 32);
      lrow[j] = lrow[j] * corr + rs;
      mrow[j] = mnew;
#pragma unroll
      for (int d = 0; d < 8; ++d) o[d][j] *= corr;
      const int prow = j + (lane >> 4) * 8;
      Ps[wave][prow][am]      = (__bf16)p0;
      Ps[wave][prow][16 + am] = (__bf16)p1;
    }
    __syncthreads();  // Ps visible (C-layout -> A-layout reshuffle via LDS)

    // O += P * V : 8 dh-subtiles; all fragments contiguous b128 pairs
    const __bf16* pr = &Ps[wave][am][0];
    const v16bf pa =
        cat16(*(const v8bf*)(pr + koff16), *(const v8bf*)(pr + koff16 + 8));
#pragma unroll
    for (int ds = 0; ds < 8; ++ds) {
      const __bf16* vr = &Vt[ds * 16 + am][0];
      const v16bf vb =
          cat16(*(const v8bf*)(vr + koff16), *(const v8bf*)(vr + koff16 + 8));
      o[ds] = wmma_bf16(pa, vb, o[ds]);
    }
  }

  // Normalize and store ctx as [B*T, D] row-major (col = h*128+dh), bf16
#pragma unroll
  for (int j = 0; j < 8; ++j) {
    const float inv = 1.0f / lrow[j];
    const int t = q0 + qrl + j + (lane >> 4) * 8;
    const size_t rowoff = (size_t)(b * SEQ + t) * DMODEL + h * DHEAD;
#pragma unroll
    for (int ds = 0; ds < 8; ++ds)
      ctx[rowoff + ds * 16 + am] = (__bf16)(o[ds][j] * inv);
  }
}

// ---------------------------------------------------------------------------
// Kernel 4: out = ctx @ w_proj.T  (bf16 x bf16 -> f32 out)
// A tile is a straight bf16 copy -> async global->LDS (ASYNCcnt) if available.
// ---------------------------------------------------------------------------
__global__ __launch_bounds__(256) void proj_gemm_kernel(
    const __bf16* __restrict__ ctx, const float* __restrict__ w_proj,
    float* __restrict__ out) {
  __shared__ __bf16 As[128][48];   // [m][k]
  __shared__ __bf16 Bs[128][48];   // [n][k] (natural w_proj row layout)

  const int tid  = threadIdx.x;
  const int lane = tid & 31;
  const int wave = tid >> 5;
  const int wm   = wave & 3;
  const int wn   = wave >> 2;
  const int row0 = blockIdx.x * 128;
  const int col0 = blockIdx.y * 128;

  const int am     = lane & 15;
  const int koff8  = (lane >> 4) * 8;
  const int koff16 = (lane >> 4) * 16;

  v8f acc[2][4];
#pragma unroll
  for (int tm = 0; tm < 2; ++tm)
#pragma unroll
    for (int tn = 0; tn < 4; ++tn)
#pragma unroll
      for (int j = 0; j < 8; ++j) acc[tm][tn][j] = 0.0f;

  for (int k0 = 0; k0 < DMODEL; k0 += 32) {
    __syncthreads();
#pragma unroll 2
    for (int idx = tid; idx < 512; idx += 256) {
      const int r  = idx >> 2;
      const int c8 = (idx & 3) * 8;
      // A already bf16: async copy straight into LDS if available
      const __bf16* asrc = ctx + (size_t)(row0 + r) * DMODEL + k0 + c8;
#ifdef HAVE_ASYNC
      async_copy16(asrc, &As[r][c8]);
#else
      *(uint4*)&As[r][c8] = *(const uint4*)asrc;
#endif
      // B: f32 -> bf16 pack through VGPRs, no transpose ([n][k] natural)
      const float* bp = w_proj + (size_t)(col0 + r) * DMODEL + k0 + c8;
      *(v8bf*)&Bs[r][c8] =
          pack8(*(const float4*)bp, *(const float4*)(bp + 4));
    }
#ifdef HAVE_ASYNC
    wait_asynccnt0();
#endif
    __syncthreads();

    v16bf a[2];
#pragma unroll
    for (int tm = 0; tm < 2; ++tm) {
      const __bf16* ar = &As[wm * 32 + tm * 16 + am][0];
      a[tm] = cat16(*(const v8bf*)(ar + koff16), *(const v8bf*)(ar + koff16 + 8));
    }
#pragma unroll
    for (int tn = 0; tn < 4; ++tn) {
      const __bf16* br = &Bs[wn * 64 + tn * 16 + am][0];
      const v16bf b =
          cat16(*(const v8bf*)(br + koff16), *(const v8bf*)(br + koff16 + 8));
#pragma unroll
      for (int tm = 0; tm < 2; ++tm)
        acc[tm][tn] = wmma_bf16(a[tm], b, acc[tm][tn]);
    }
  }

#pragma unroll
  for (int tm = 0; tm < 2; ++tm)
#pragma unroll
    for (int tn = 0; tn < 4; ++tn)
#pragma unroll
      for (int j = 0; j < 8; ++j) {
        const int r = row0 + wm * 32 + tm * 16 + j + (lane >> 4) * 8;
        const int c = col0 + wn * 64 + tn * 16 + am;
        out[(size_t)r * DMODEL + c] = acc[tm][tn][j];
      }
}

// ---------------------------------------------------------------------------
extern "C" void kernel_launch(void* const* d_in, const int* in_sizes, int n_in,
                              void* d_out, int out_size, void* d_ws,
                              size_t ws_size, hipStream_t stream) {
  const float* x      = (const float*)d_in[0];
  const float* w_qkv  = (const float*)d_in[1];
  const float* w_proj = (const float*)d_in[2];
  const int*   seqoff = (const int*)d_in[3];
  float* out = (float*)d_out;

  __bf16* qbuf = (__bf16*)d_ws;
  __bf16* kbuf = qbuf + HEAD_ELEMS;
  __bf16* vbuf = kbuf + HEAD_ELEMS;
  __bf16* ctx  = vbuf + HEAD_ELEMS;

  // 1) QKV projection (bf16 WMMA, f32 accum)
  qkv_gemm_kernel<<<dim3(NROWS / 128, NQKV / 128), 256, 0, stream>>>(
      x, w_qkv, qbuf, kbuf, vbuf);

  // 2) RoPE on q,k
  {
    const int total = BATCH * SEQ * NHEAD * 64;
    rope_kernel<<<total / 256, 256, 0, stream>>>(qbuf, kbuf, seqoff);
  }

  // 3) Causal flash attention
  attn_kernel<<<dim3(SEQ / 128, BATCH * NHEAD), 256, 0, stream>>>(
      qbuf, kbuf, vbuf, ctx);

  // 4) Output projection
  proj_gemm_kernel<<<dim3(NROWS / 128, DMODEL / 128), 256, 0, stream>>>(
      ctx, w_proj, out);
}
